// MultiHeadAttention_44341242364052
// MI455X (gfx1250) — matrix operand order repaired and checked by reading, not compiled, not run on hardware
//
#include <hip/hip_runtime.h>
#include <cstddef>

// ---------------------------------------------------------------------------
// MHA for B=2, S=2048, D=1024, H=16, DK=64 on gfx1250 (CDNA5, wave32, WMMA).
// Pipeline: 3x projection GEMM (f32 in -> f16 out, double-buffered LDS),
// flash attention (f16, async LDS-DMA for K tiles), output GEMM (f16 -> f32).
// All matmuls via v_wmma_f32_16x16x32_f16.
// ---------------------------------------------------------------------------

typedef _Float16 half8  __attribute__((ext_vector_type(8)));
typedef _Float16 half16 __attribute__((ext_vector_type(16)));
typedef float    floatx4 __attribute__((ext_vector_type(4)));
typedef float    floatx8 __attribute__((ext_vector_type(8)));

static constexpr int Bb = 2;
static constexpr int Ss = 2048;
static constexpr int Dd = 1024;
static constexpr int Hh = 16;
static constexpr int DKk = 64;

__device__ __forceinline__ half16 make_frag(half8 lo, half8 hi) {
  half16 r;
#pragma unroll
  for (int i = 0; i < 8; ++i) { r[i] = lo[i]; r[i + 8] = hi[i]; }
  return r;
}

// Async LDS-DMA: copy 16 bytes per lane from global to LDS (ASYNCcnt-tracked).
__device__ __forceinline__ void async_load_b128(const void* gptr, void* lptr) {
  const unsigned lds = (unsigned)(unsigned long long)lptr;  // low 32b = LDS addr
  asm volatile("global_load_async_to_lds_b128 %0, %1, off"
               :: "v"(lds), "v"(gptr)
               : "memory");
}
__device__ __forceinline__ void wait_async_zero() {
  asm volatile("s_wait_asynccnt 0x0" ::: "memory");
}

// ---------------------------------------------------------------------------
// Tiled WMMA GEMM:  C[M,N] = A[M,K] @ W[N,K]^T + bias[N]
// Block tile 128x128, K-step 32, 256 threads = 8 waves (4 M x 2 N),
// each wave computes 32x64 via 2x4 wmma accumulators. Ping-pong LDS tiles,
// next K-slab prefetched into registers while the matrix pipe consumes LDS.
// ---------------------------------------------------------------------------
template <typename TA, typename TOut>
__global__ __launch_bounds__(256) void gemm_wmma_kernel(
    const TA* __restrict__ A, const float* __restrict__ W,
    const float* __restrict__ bias, TOut* __restrict__ C,
    int M, int N, int K) {
  constexpr int BM = 128, BN = 128, BK = 32;
  constexpr int LDA = BK + 8;  // pad (40 halves = 80B) to dodge bank conflicts
  __shared__ _Float16 As[2][BM * LDA];
  __shared__ _Float16 Ws[2][BN * LDA];

  const int tid  = threadIdx.x;
  const int lane = tid & 31;
  const int wave = tid >> 5;
  const int waveM = (wave & 3) * 32;
  const int waveN = (wave >> 2) * 64;
  const int blockM = blockIdx.x * BM;
  const int blockN = blockIdx.y * BN;

  floatx8 acc[2][4];
#pragma unroll
  for (int i = 0; i < 2; ++i)
#pragma unroll
    for (int j = 0; j < 4; ++j) acc[i][j] = {};

  // staging: 256 threads cover 128 rows x two 16-wide K segments
  const int sRow = tid >> 1;
  const int sK0  = (tid & 1) * 16;

  const int laneN   = lane & 15;
  const int kg      = lane >> 4;    // K-group within fragment
  const int halfSel = kg;           // C-layout row half

  // load one thread's 16-element K segment of the A tile into registers (f16)
  auto loadA = [&](int k0, half8& h0, half8& h1) {
    const TA* src = A + (size_t)(blockM + sRow) * K + k0 + sK0;
    if constexpr (sizeof(TA) == 4) {
      floatx4 v0 = *(const floatx4*)(src);
      floatx4 v1 = *(const floatx4*)(src + 4);
      floatx4 v2 = *(const floatx4*)(src + 8);
      floatx4 v3 = *(const floatx4*)(src + 12);
#pragma unroll
      for (int i = 0; i < 4; ++i) {
        h0[i] = (_Float16)v0[i]; h0[i + 4] = (_Float16)v1[i];
        h1[i] = (_Float16)v2[i]; h1[i + 4] = (_Float16)v3[i];
      }
    } else {
      h0 = *(const half8*)(src);
      h1 = *(const half8*)(src + 8);
    }
  };
  auto loadW = [&](int k0, half8& h0, half8& h1) {
    const float* src = W + (size_t)(blockN + sRow) * K + k0 + sK0;
    floatx4 v0 = *(const floatx4*)(src);
    floatx4 v1 = *(const floatx4*)(src + 4);
    floatx4 v2 = *(const floatx4*)(src + 8);
    floatx4 v3 = *(const floatx4*)(src + 12);
#pragma unroll
    for (int i = 0; i < 4; ++i) {
      h0[i] = (_Float16)v0[i]; h0[i + 4] = (_Float16)v1[i];
      h1[i] = (_Float16)v2[i]; h1[i + 4] = (_Float16)v3[i];
    }
  };
  auto stash = [&](_Float16* dst, half8 h0, half8 h1) {
    *(half8*)(dst)     = h0;
    *(half8*)(dst + 8) = h1;
  };

  // prologue: stage slab 0 into buffer 0
  {
    half8 a0, a1, w0, w1;
    loadA(0, a0, a1);
    loadW(0, w0, w1);
    stash(&As[0][sRow * LDA + sK0], a0, a1);
    stash(&Ws[0][sRow * LDA + sK0], w0, w1);
  }
  __syncthreads();

  const int nIter = K / BK;
  for (int it = 0; it < nIter; ++it) {
    const int cur = it & 1;
    const int nxt = cur ^ 1;
    const int k0n = (it + 1) * BK;
    const bool hasNext = k0n < K;

    // prefetch next slab into registers (overlaps with WMMA below)
    half8 na0, na1, nw0, nw1;
    if (hasNext) {
      loadA(k0n, na0, na1);
      loadW(k0n, nw0, nw1);
    }

    // A fragments: row per lane (laneN), K striped: v0..3 = K kg*8..+7,
    // v4..7 = K 16+kg*8..+7   (ISA 16-bit A 16x32 layout)
    half16 afrag[2];
#pragma unroll
    for (int i = 0; i < 2; ++i) {
      const _Float16* p = &As[cur][(waveM + i * 16 + laneN) * LDA];
      afrag[i] = make_frag(*(const half8*)(p + kg * 8),
                           *(const half8*)(p + 16 + kg * 8));
    }
    // B fragments: column (output N = weight row) per lane, K striped:
    // v0..3 = K kg*8..+7, v4..7 = K 16+kg*8..+7
    half16 bfrag[4];
#pragma unroll
    for (int j = 0; j < 4; ++j) {
      const _Float16* p = &Ws[cur][(waveN + j * 16 + laneN) * LDA + kg * 8];
      bfrag[j] = make_frag(*(const half8*)(p), *(const half8*)(p + 16));
    }
#pragma unroll
    for (int i = 0; i < 2; ++i)
#pragma unroll
      for (int j = 0; j < 4; ++j)
        acc[i][j] = __builtin_amdgcn_wmma_f32_16x16x32_f16(
            false, afrag[i], false, bfrag[j], (short)0, acc[i][j], false, false);

    if (hasNext) {
      stash(&As[nxt][sRow * LDA + sK0], na0, na1);
      stash(&Ws[nxt][sRow * LDA + sK0], nw0, nw1);
    }
    __syncthreads();
  }

  // epilogue: C layout = row per accumulator element, col per lane
#pragma unroll
  for (int j = 0; j < 4; ++j) {
    const int col = blockN + waveN + j * 16 + laneN;
    const float bv = bias[col];
#pragma unroll
    for (int i = 0; i < 2; ++i) {
      const int rowBase = blockM + waveM + i * 16 + halfSel * 8;
#pragma unroll
      for (int r = 0; r < 8; ++r) {
        const float val = acc[i][j][r] + bv;
        C[(size_t)(rowBase + r) * N + col] = (TOut)val;
      }
    }
  }
}

// ---------------------------------------------------------------------------
// Flash attention (causal), f16 in/out, f32 accumulate.
// One block (4 waves, 128 threads) per (b, h, 64-query tile).
// Each wave owns 16 query rows; key tiles of 32; DK = 64.
// K tiles staged via async LDS-DMA (ASYNCcnt); V staged transposed.
// ---------------------------------------------------------------------------
__global__ __launch_bounds__(128) void attn_wmma_kernel(
    const _Float16* __restrict__ Qb, const _Float16* __restrict__ Kb,
    const _Float16* __restrict__ Vb, _Float16* __restrict__ Ob) {
  constexpr int QT = 64, KT = 32;
  __shared__ _Float16 Ks[KT][DKk];        // 32 keys x 64 dk
  __shared__ _Float16 Vts[DKk][KT + 8];   // transposed V: [dk][key], padded
  __shared__ _Float16 Ps[4][16][KT];      // per-wave P transpose scratch

  const int tid  = threadIdx.x;
  const int lane = tid & 31;
  const int wave = tid >> 5;
  const int laneN = lane & 15;
  const int kg    = lane >> 4;
  const int halfSel = kg;

  int bx = blockIdx.x;                 // B * H * (S/QT)
  const int qTile = bx % (Ss / QT); bx /= (Ss / QT);
  const int h = bx % Hh;
  const int b = bx / Hh;
  const int q0 = qTile * QT;

  const size_t base = (size_t)b * Ss * Dd + (size_t)h * DKk;

  // Q fragments: two K-chunks of 32 covering DK=64, loaded straight from global
  half16 qfrag[2];
  {
    const _Float16* qp = Qb + base + (size_t)(q0 + wave * 16 + laneN) * Dd;
#pragma unroll
    for (int j = 0; j < 2; ++j) {
      const int c0 = j * 32 + kg * 8;
      qfrag[j] = make_frag(*(const half8*)(qp + c0),
                           *(const half8*)(qp + 16 + c0));
    }
  }

  floatx8 o[4];
#pragma unroll
  for (int t = 0; t < 4; ++t) o[t] = {};
  float mrow[8], lrow[8];
#pragma unroll
  for (int r = 0; r < 8; ++r) { mrow[r] = -__builtin_inff(); lrow[r] = 0.0f; }

  // per-thread staging coordinates: 128 threads x 16 halves
  const int sKey = tid >> 2;
  const int sC0  = (tid & 3) * 16;

  const int ktEnd = q0 + QT;  // causal upper bound on keys
  for (int kt = 0; kt < ktEnd; kt += KT) {
    {  // stage K tile via async LDS-DMA
      const _Float16* src = Kb + base + (size_t)(kt + sKey) * Dd + sC0;
      async_load_b128(src,     &Ks[sKey][sC0]);
      async_load_b128(src + 8, &Ks[sKey][sC0 + 8]);
    }
    {  // stage V tile transposed [dk][key]
      const _Float16* src = Vb + base + (size_t)(kt + sKey) * Dd + sC0;
      half8 v0 = *(const half8*)(src);
      half8 v1 = *(const half8*)(src + 8);
#pragma unroll
      for (int i = 0; i < 8; ++i) {
        Vts[sC0 + i][sKey]     = v0[i];
        Vts[sC0 + 8 + i][sKey] = v1[i];
      }
    }
    if (kt + KT < ktEnd) {  // hint next tiles toward the caches
      __builtin_prefetch(Kb + base + (size_t)(kt + KT + sKey) * Dd + sC0, 0, 3);
      __builtin_prefetch(Vb + base + (size_t)(kt + KT + sKey) * Dd + sC0, 0, 3);
    }
    wait_async_zero();
    __syncthreads();

    // scores: this wave's 16 queries x 32 keys, scaled by 1/sqrt(DK)
    floatx8 s[2];
#pragma unroll
    for (int n = 0; n < 2; ++n) {
      s[n] = {};
#pragma unroll
      for (int j = 0; j < 2; ++j) {
        const _Float16* p = &Ks[n * 16 + laneN][j * 32 + kg * 8];
        half16 bf = make_frag(*(const half8*)p, *(const half8*)(p + 16));
        s[n] = __builtin_amdgcn_wmma_f32_16x16x32_f16(
            false, qfrag[j], false, bf, (short)0, s[n], false, false);
      }
#pragma unroll
      for (int r = 0; r < 8; ++r) s[n][r] *= 0.125f;
    }

    // causal mask (only tiles straddling the diagonal)
    const int qRowBase = q0 + wave * 16 + halfSel * 8;
    if (kt + KT - 1 > q0 + wave * 16) {
#pragma unroll
      for (int n = 0; n < 2; ++n) {
        const int col = kt + n * 16 + laneN;
#pragma unroll
        for (int r = 0; r < 8; ++r)
          if (col > qRowBase + r) s[n][r] = -__builtin_inff();
      }
    }

    // online softmax: row reductions across 16 lanes of each half-wave
    float pvals[2][8];
    float corr[8];
#pragma unroll
    for (int r = 0; r < 8; ++r) {
      float mx = fmaxf(s[0][r], s[1][r]);
#pragma unroll
      for (int off = 1; off < 16; off <<= 1)
        mx = fmaxf(mx, __shfl_xor(mx, off, 32));
      const float mnew = fmaxf(mrow[r], mx);
      const float c    = __expf(mrow[r] - mnew);
      const float p0   = __expf(s[0][r] - mnew);
      const float p1   = __expf(s[1][r] - mnew);
      float sum = p0 + p1;
#pragma unroll
      for (int off = 1; off < 16; off <<= 1)
        sum += __shfl_xor(sum, off, 32);
      lrow[r] = lrow[r] * c + sum;
      mrow[r] = mnew;
      corr[r] = c;
      pvals[0][r] = p0;
      pvals[1][r] = p1;
    }
#pragma unroll
    for (int t = 0; t < 4; ++t)
#pragma unroll
      for (int r = 0; r < 8; ++r) o[t][r] *= corr[r];

    // transpose P (C layout -> A fragment layout) through per-wave LDS.
    // Same-wave LDS ops are in-order; no barrier needed.
#pragma unroll
    for (int n = 0; n < 2; ++n)
#pragma unroll
      for (int r = 0; r < 8; ++r)
        Ps[wave][halfSel * 8 + r][n * 16 + laneN] = (_Float16)pvals[n][r];

    const _Float16* pp = &Ps[wave][laneN][0];
    half16 pfrag = make_frag(*(const half8*)(pp + kg * 8),
                             *(const half8*)(pp + 16 + kg * 8));

    // O += P @ V  (B fragments from transposed V tile)
#pragma unroll
    for (int t = 0; t < 4; ++t) {
      const _Float16* vp = &Vts[t * 16 + laneN][kg * 8];
      half16 vf = make_frag(*(const half8*)vp, *(const half8*)(vp + 16));
      o[t] = __builtin_amdgcn_wmma_f32_16x16x32_f16(
          false, pfrag, false, vf, (short)0, o[t], false, false);
    }
    __syncthreads();
  }

  // epilogue: normalize and write merged-head output (B,S,D) as f16
  const int rowOutBase = q0 + wave * 16 + halfSel * 8;
#pragma unroll
  for (int r = 0; r < 8; ++r) {
    const float inv = 1.0f / lrow[r];
    _Float16* dst = Ob + base + (size_t)(rowOutBase + r) * Dd;
#pragma unroll
    for (int t = 0; t < 4; ++t)
      dst[t * 16 + laneN] = (_Float16)(o[t][r] * inv);
  }
}

// ---------------------------------------------------------------------------
// Launch
// ---------------------------------------------------------------------------
extern "C" void kernel_launch(void* const* d_in, const int* in_sizes, int n_in,
                              void* d_out, int out_size, void* d_ws,
                              size_t ws_size, hipStream_t stream) {
  (void)in_sizes; (void)n_in; (void)out_size; (void)ws_size;
  const float* q  = (const float*)d_in[0];
  const float* k  = (const float*)d_in[1];
  const float* v  = (const float*)d_in[2];
  // d_in[3] = causal mask: reproduced analytically in the attention kernel
  const float* wq = (const float*)d_in[4];
  const float* bq = (const float*)d_in[5];
  const float* wk = (const float*)d_in[6];
  const float* bk = (const float*)d_in[7];
  const float* wv = (const float*)d_in[8];
  const float* bv = (const float*)d_in[9];
  const float* wo = (const float*)d_in[10];
  const float* bo = (const float*)d_in[11];
  float* out = (float*)d_out;

  const int M = Bb * Ss;   // 4096
  const int N = Dd;        // 1024
  const int K = Dd;        // 1024
  const size_t elems = (size_t)Bb * Ss * Dd;

  _Float16* Qb = (_Float16*)d_ws;
  _Float16* Kb = Qb + elems;
  _Float16* Vb = Kb + elems;
  _Float16* Ob = Vb + elems;

  dim3 gGemm(M / 128, N / 128);  // 32 x 8
  dim3 bGemm(256);

  gemm_wmma_kernel<float, _Float16><<<gGemm, bGemm, 0, stream>>>(q, wq, bq, Qb, M, N, K);
  gemm_wmma_kernel<float, _Float16><<<gGemm, bGemm, 0, stream>>>(k, wk, bk, Kb, M, N, K);
  gemm_wmma_kernel<float, _Float16><<<gGemm, bGemm, 0, stream>>>(v, wv, bv, Vb, M, N, K);

  const int nAttnBlocks = Bb * Hh * (Ss / 64);  // 1024
  attn_wmma_kernel<<<nAttnBlocks, 128, 0, stream>>>(Qb, Kb, Vb, Ob);

  gemm_wmma_kernel<_Float16, float><<<gGemm, bGemm, 0, stream>>>(Ob, wo, bo, out, M, N, K);
}
